// MAGNAKGlayer_61581241090293
// MI455X (gfx1250) — compile-verified, matrix operand order checked by reading, not run on hardware
//
#include <hip/hip_runtime.h>
#include <hip/hip_bf16.h>

// ---------------- problem constants ----------------
#define NN   50000
#define EE   800000
#define RR   500
#define RPAD 512
#define DD   256
#define HH   8
#define AD   32
#define HOPS 3
#define ALPHA 0.15f
#define SLOPE 0.2f

typedef __attribute__((ext_vector_type(16))) __bf16          bf16x16;
typedef __attribute__((ext_vector_type(16))) unsigned short  u16x16;
typedef __attribute__((ext_vector_type(8)))  unsigned short  u16x8;
typedef __attribute__((ext_vector_type(8)))  float           v8f;

__device__ __forceinline__ unsigned short f2bf(float f) {
    unsigned u = __float_as_uint(f);
    unsigned r = u + 0x7fffu + ((u >> 16) & 1u);   // round-to-nearest-even
    return (unsigned short)(r >> 16);
}

// ---------------- fill ----------------
__global__ void fill_u32_kernel(unsigned* __restrict__ p, unsigned val, size_t n) {
    size_t i = (size_t)blockIdx.x * blockDim.x + threadIdx.x;
    if (i < n) p[i] = val;
}

// ---------------- weight transpose + f32->bf16 :  W[K,Nc] -> Wt[Nc,K] ----------------
__global__ void wtr_kernel(const float* __restrict__ W, unsigned short* __restrict__ Wt,
                           int K, int Nc) {
    int idx = blockIdx.x * 256 + threadIdx.x;
    if (idx >= K * Nc) return;
    int k = idx / Nc, n = idx % Nc;
    Wt[(size_t)n * K + k] = f2bf(W[idx]);
}

// ---------------- LayerNorm (D=256) -> bf16; pads rows [M,Mpad) with zeros ----------
__global__ __launch_bounds__(256)
void ln_bf16_kernel(const float* __restrict__ x, const float* __restrict__ g,
                    const float* __restrict__ b, unsigned short* __restrict__ out,
                    int M, int Mpad) {
    int row = blockIdx.x;
    int t = threadIdx.x;
    if (row >= Mpad) return;
    size_t base = (size_t)row * DD;
    if (row >= M) { out[base + t] = 0; return; }
    float v = x[base + t];
    __shared__ float s1[256], s2[256];
    s1[t] = v; s2[t] = v * v;
    __syncthreads();
    for (int off = 128; off; off >>= 1) {
        if (t < off) { s1[t] += s1[t + off]; s2[t] += s2[t + off]; }
        __syncthreads();
    }
    float mu  = s1[0] * (1.f / 256.f);
    float var = s2[0] * (1.f / 256.f) - mu * mu;
    float inv = rsqrtf(var + 1e-5f);
    out[base + t] = f2bf(g[t] * (v - mu) * inv + b[t]);
}

// ---------------- WMMA GEMM: C[M,Nc] = A[M,K](bf16) x Bt[Nc,K](bf16)^T ---------------
// Each wave computes a 16x64 slab: one A fragment reused against 4 B fragments
// (4 accumulators, 4 x v_wmma_f32_16x16x32_bf16 per 32-wide k-step).
// Requires (Nc/16) % 4 == 0 and M % 16 == 0 (true for all call sites).
// MODE 0: store f32            1: store tanh f32
// MODE 2: + resid, f32         3: + bias, relu, bf16
// MODE 4: + bias + resid, f32
template <int MODE>
__global__ __launch_bounds__(256)
void wmma_gemm_kernel(const unsigned short* __restrict__ A,
                      const unsigned short* __restrict__ Bt,
                      const float* __restrict__ bias,
                      const float* __restrict__ resid,
                      float* __restrict__ Cf,
                      unsigned short* __restrict__ Cb,
                      int M, int K, int Nc) {
    int groupsN = Nc >> 6;                       // groups of 4 N-tiles
    int totalGroups = (M >> 4) * groupsN;
    int wid = blockIdx.x * 8 + (threadIdx.x >> 5);
    if (wid >= totalGroups) return;              // wave-uniform branch
    int lane = threadIdx.x & 31;
    int tm = wid / groupsN, tg = wid % groupsN;
    int l15  = lane & 15;
    int hi16 = (lane >> 4) & 1;

    const unsigned short* Arow = A + (size_t)(tm * 16 + l15) * K;
    const unsigned short* Brow0 = Bt + (size_t)(tg * 64 + 0  + l15) * K;
    const unsigned short* Brow1 = Bt + (size_t)(tg * 64 + 16 + l15) * K;
    const unsigned short* Brow2 = Bt + (size_t)(tg * 64 + 32 + l15) * K;
    const unsigned short* Brow3 = Bt + (size_t)(tg * 64 + 48 + l15) * K;

    v8f acc0 = {}, acc1 = {}, acc2 = {}, acc3 = {};
    for (int k0 = 0; k0 < K; k0 += 32) {
        int klo = k0 + hi16 * 8;        // VGPRs 0-3 : K=0..7 / 8..15 per lane half
        int khi = k0 + 16 + hi16 * 8;   // VGPRs 4-7 : K=16..23 / 24..31

        u16x8 alo = *(const u16x8*)(Arow + klo);
        u16x8 ahi = *(const u16x8*)(Arow + khi);
        u16x16 ap;
#pragma unroll
        for (int i = 0; i < 8; ++i) { ap[i] = alo[i]; ap[8 + i] = ahi[i]; }
        bf16x16 av = __builtin_bit_cast(bf16x16, ap);

        u16x8 b0l = *(const u16x8*)(Brow0 + klo), b0h = *(const u16x8*)(Brow0 + khi);
        u16x8 b1l = *(const u16x8*)(Brow1 + klo), b1h = *(const u16x8*)(Brow1 + khi);
        u16x8 b2l = *(const u16x8*)(Brow2 + klo), b2h = *(const u16x8*)(Brow2 + khi);
        u16x8 b3l = *(const u16x8*)(Brow3 + klo), b3h = *(const u16x8*)(Brow3 + khi);
        u16x16 bp0, bp1, bp2, bp3;
#pragma unroll
        for (int i = 0; i < 8; ++i) {
            bp0[i] = b0l[i]; bp0[8 + i] = b0h[i];
            bp1[i] = b1l[i]; bp1[8 + i] = b1h[i];
            bp2[i] = b2l[i]; bp2[8 + i] = b2h[i];
            bp3[i] = b3l[i]; bp3[8 + i] = b3h[i];
        }
        acc0 = __builtin_amdgcn_wmma_f32_16x16x32_bf16(false, av, false,
                 __builtin_bit_cast(bf16x16, bp0), (short)0, acc0, false, false);
        acc1 = __builtin_amdgcn_wmma_f32_16x16x32_bf16(false, av, false,
                 __builtin_bit_cast(bf16x16, bp1), (short)0, acc1, false, false);
        acc2 = __builtin_amdgcn_wmma_f32_16x16x32_bf16(false, av, false,
                 __builtin_bit_cast(bf16x16, bp2), (short)0, acc2, false, false);
        acc3 = __builtin_amdgcn_wmma_f32_16x16x32_bf16(false, av, false,
                 __builtin_bit_cast(bf16x16, bp3), (short)0, acc3, false, false);
    }

    // C layout: VGPR r -> row M=r (+8 for hi lanes), col N = lane&15
#pragma unroll
    for (int j = 0; j < 4; ++j) {
        v8f acc = (j == 0) ? acc0 : (j == 1) ? acc1 : (j == 2) ? acc2 : acc3;
        int ocol = tg * 64 + j * 16 + l15;
#pragma unroll
        for (int r = 0; r < 8; ++r) {
            int orow = tm * 16 + r + hi16 * 8;
            size_t idx = (size_t)orow * Nc + ocol;
            float v = acc[r];
            if (MODE == 0)      Cf[idx] = v;
            else if (MODE == 1) Cf[idx] = tanhf(v);
            else if (MODE == 2) Cf[idx] = v + resid[idx];
            else if (MODE == 3) { v += bias[ocol]; Cb[idx] = f2bf(v > 0.f ? v : 0.f); }
            else                Cf[idx] = v + bias[ocol] + resid[idx];
        }
    }
}

// ---------------- attention reduce: out[m,h] = sum_ad proj[m,h*32+ad]*attn[h,ad] -----
__global__ __launch_bounds__(256)
void attn_reduce_kernel(const float* __restrict__ proj, const float* __restrict__ attn,
                        float* __restrict__ out, int M) {
    int row = blockIdx.x;
    if (row >= M) return;
    int t = threadIdx.x, hd = t >> 5, ad = t & 31;
    float v = proj[(size_t)row * DD + t] * attn[hd * AD + ad];
    for (int off = 16; off; off >>= 1) v += __shfl_down(v, off, 32);
    if (ad == 0) out[(size_t)row * HH + hd] = v;
}

// ---------------- edge softmax ----------------
__device__ __forceinline__ unsigned enc_f32(float f) {
    unsigned u = __float_as_uint(f);
    return (u & 0x80000000u) ? ~u : (u | 0x80000000u);
}
__device__ __forceinline__ float dec_f32(unsigned u) {
    return (u & 0x80000000u) ? __uint_as_float(u & 0x7fffffffu) : __uint_as_float(~u);
}

__global__ void edge_score_kernel(const int* __restrict__ src, const int* __restrict__ dst,
                                  const int* __restrict__ eid,
                                  const float* __restrict__ eh, const float* __restrict__ et,
                                  const float* __restrict__ er,
                                  float* __restrict__ scores, unsigned* __restrict__ menc) {
    int i = blockIdx.x * 256 + threadIdx.x;
    if (i >= EE * HH) return;
    int e = i >> 3, h = i & 7;
    float s = eh[(size_t)src[e] * HH + h] + et[(size_t)dst[e] * HH + h]
            + er[(size_t)eid[e] * HH + h];
    s = s > 0.f ? s : SLOPE * s;
    scores[i] = s;
    atomicMax(&menc[(size_t)dst[e] * HH + h], enc_f32(s));
}

__global__ void edge_exp_kernel(const int* __restrict__ dst, float* __restrict__ scores,
                                const unsigned* __restrict__ menc, float* __restrict__ den) {
    int i = blockIdx.x * 256 + threadIdx.x;
    if (i >= EE * HH) return;
    int e = i >> 3, h = i & 7;
    float m = dec_f32(menc[(size_t)dst[e] * HH + h]);
    float ex = __expf(scores[i] - m);
    scores[i] = ex;
    atomicAdd(&den[(size_t)dst[e] * HH + h], ex);
}

__global__ void edge_norm_kernel(const int* __restrict__ dst, float* __restrict__ scores,
                                 const float* __restrict__ den) {
    int i = blockIdx.x * 256 + threadIdx.x;
    if (i >= EE * HH) return;
    int e = i >> 3, h = i & 7;
    scores[i] = scores[i] / den[(size_t)dst[e] * HH + h];
}

// ---------------- propagation hop: accum[dst] += feat[src] * a ----------------
__global__ __launch_bounds__(256)
void hop_kernel(const int* __restrict__ src, const int* __restrict__ dst,
                const float* __restrict__ a, const float* __restrict__ featIn,
                float* __restrict__ accum) {
    int e = blockIdx.x;
    int c = threadIdx.x;
    float v = featIn[(size_t)src[e] * DD + c] * a[(size_t)e * HH + (c >> 5)];
    atomicAdd(&accum[(size_t)dst[e] * DD + c], v);
}

__global__ void combine_kernel(float* __restrict__ featOut, float* __restrict__ accum,
                               const float* __restrict__ feat0, size_t n) {
    size_t i = (size_t)blockIdx.x * blockDim.x + threadIdx.x;
    if (i < n) {
        featOut[i] = (1.f - ALPHA) * accum[i] + ALPHA * feat0[i];
        accum[i] = 0.f;                       // re-zero for next hop / next call
    }
}

__global__ void f32_to_bf16_kernel(const float* __restrict__ x,
                                   unsigned short* __restrict__ y, size_t n) {
    size_t i = (size_t)blockIdx.x * blockDim.x + threadIdx.x;
    if (i < n) y[i] = f2bf(x[i]);
}

// ---------------- launch ----------------
static inline int cdiv(long long a, long long b) { return (int)((a + b - 1) / b); }

extern "C" void kernel_launch(void* const* d_in, const int* in_sizes, int n_in,
                              void* d_out, int out_size, void* d_ws, size_t ws_size,
                              hipStream_t stream) {
    const float* ent   = (const float*)d_in[0];
    const float* rel   = (const float*)d_in[1];
    const int*   src   = (const int*)d_in[2];
    const int*   dst   = (const int*)d_in[3];
    const int*   eid   = (const int*)d_in[4];
    const float* g_e   = (const float*)d_in[5];
    const float* b_e   = (const float*)d_in[6];
    const float* g_r   = (const float*)d_in[7];
    const float* b_r   = (const float*)d_in[8];
    const float* g_ff  = (const float*)d_in[9];
    const float* b_ff  = (const float*)d_in[10];
    const float* Whead = (const float*)d_in[11];
    const float* Wtail = (const float*)d_in[12];
    const float* Went  = (const float*)d_in[13];
    const float* Wrel  = (const float*)d_in[14];
    const float* Wout  = (const float*)d_in[15];
    const float* attnh = (const float*)d_in[16];
    const float* attnt = (const float*)d_in[17];
    const float* attnr = (const float*)d_in[18];
    const float* w1    = (const float*)d_in[19];
    const float* b1    = (const float*)d_in[20];
    const float* w2    = (const float*)d_in[21];
    const float* b2    = (const float*)d_in[22];
    float* outp = (float*)d_out;

    // ---- workspace carve-up ----
    char* p = (char*)d_ws;
    auto alloc = [&](size_t bytes) { char* r = p; p += (bytes + 255) & ~(size_t)255; return r; };
    unsigned short* hb     = (unsigned short*)alloc((size_t)NN * DD * 2);
    unsigned short* relb   = (unsigned short*)alloc((size_t)RPAD * DD * 2);
    unsigned short* WtH    = (unsigned short*)alloc((size_t)DD * DD * 2);
    unsigned short* WtT    = (unsigned short*)alloc((size_t)DD * DD * 2);
    unsigned short* WtE    = (unsigned short*)alloc((size_t)DD * DD * 2);
    unsigned short* WtR    = (unsigned short*)alloc((size_t)DD * DD * 2);
    unsigned short* WtO    = (unsigned short*)alloc((size_t)DD * DD * 2);
    unsigned short* Wt1    = (unsigned short*)alloc((size_t)DD * 4 * DD * 2);
    unsigned short* Wt2    = (unsigned short*)alloc((size_t)DD * 4 * DD * 2);
    float* projA           = (float*)alloc((size_t)NN * DD * 4);
    float* projR           = (float*)alloc((size_t)RPAD * DD * 4);
    float* eh              = (float*)alloc((size_t)NN * HH * 4);
    float* et              = (float*)alloc((size_t)NN * HH * 4);
    float* er              = (float*)alloc((size_t)RPAD * HH * 4);
    float* scores          = (float*)alloc((size_t)EE * HH * 4);
    unsigned* menc         = (unsigned*)alloc((size_t)NN * HH * 4);
    float* den             = (float*)alloc((size_t)NN * HH * 4);
    float* feat0           = (float*)alloc((size_t)NN * DD * 4);
    float* featA           = (float*)alloc((size_t)NN * DD * 4);
    float* featB           = (float*)alloc((size_t)NN * DD * 4);
    unsigned short* featAb = (unsigned short*)alloc((size_t)NN * DD * 2);
    float* rst             = (float*)alloc((size_t)NN * DD * 4);
    unsigned short* xnb    = (unsigned short*)alloc((size_t)NN * DD * 2);
    unsigned short* midb   = (unsigned short*)alloc((size_t)NN * 4 * DD * 2);

    // ---- weights -> bf16, transposed to [out,in] ----
    wtr_kernel<<<cdiv(DD * DD, 256), 256, 0, stream>>>(Whead, WtH, DD, DD);
    wtr_kernel<<<cdiv(DD * DD, 256), 256, 0, stream>>>(Wtail, WtT, DD, DD);
    wtr_kernel<<<cdiv(DD * DD, 256), 256, 0, stream>>>(Went,  WtE, DD, DD);
    wtr_kernel<<<cdiv(DD * DD, 256), 256, 0, stream>>>(Wrel,  WtR, DD, DD);
    wtr_kernel<<<cdiv(DD * DD, 256), 256, 0, stream>>>(Wout,  WtO, DD, DD);
    wtr_kernel<<<cdiv(DD * 4 * DD, 256), 256, 0, stream>>>(w1, Wt1, DD, 4 * DD);
    wtr_kernel<<<cdiv(DD * 4 * DD, 256), 256, 0, stream>>>(w2, Wt2, 4 * DD, DD);

    // ---- LayerNorms -> bf16 ----
    ln_bf16_kernel<<<NN, 256, 0, stream>>>(ent, g_e, b_e, hb, NN, NN);
    ln_bf16_kernel<<<RPAD, 256, 0, stream>>>(rel, g_r, b_r, relb, RR, RPAD);

    // ---- relation path: tanh(h_r @ W_rel) then reduce with attn_r ----
    {
        int groups = (RPAD / 16) * (DD / 64);
        wmma_gemm_kernel<1><<<cdiv(groups, 8), 256, 0, stream>>>(
            relb, WtR, nullptr, nullptr, projR, nullptr, RPAD, DD, DD);
        attn_reduce_kernel<<<RR, 256, 0, stream>>>(projR, attnr, er, RR);
    }

    // ---- entity projections ----
    int groupsNN = (NN / 16) * (DD / 64);
    wmma_gemm_kernel<1><<<cdiv(groupsNN, 8), 256, 0, stream>>>(
        hb, WtH, nullptr, nullptr, projA, nullptr, NN, DD, DD);
    attn_reduce_kernel<<<NN, 256, 0, stream>>>(projA, attnh, eh, NN);
    wmma_gemm_kernel<1><<<cdiv(groupsNN, 8), 256, 0, stream>>>(
        hb, WtT, nullptr, nullptr, projA, nullptr, NN, DD, DD);
    attn_reduce_kernel<<<NN, 256, 0, stream>>>(projA, attnt, et, NN);
    wmma_gemm_kernel<0><<<cdiv(groupsNN, 8), 256, 0, stream>>>(
        hb, WtE, nullptr, nullptr, feat0, nullptr, NN, DD, DD);

    // ---- edge softmax ----
    size_t nNH = (size_t)NN * HH;
    fill_u32_kernel<<<cdiv(nNH, 256), 256, 0, stream>>>(menc, 0u, nNH);          // -> -inf
    fill_u32_kernel<<<cdiv(nNH, 256), 256, 0, stream>>>((unsigned*)den, 0u, nNH);
    size_t nND = (size_t)NN * DD;
    fill_u32_kernel<<<cdiv(nND, 256), 256, 0, stream>>>((unsigned*)featB, 0u, nND);

    int eg = cdiv((long long)EE * HH, 256);
    edge_score_kernel<<<eg, 256, 0, stream>>>(src, dst, eid, eh, et, er, scores, menc);
    edge_exp_kernel<<<eg, 256, 0, stream>>>(dst, scores, menc, den);
    edge_norm_kernel<<<eg, 256, 0, stream>>>(dst, scores, den);

    // ---- 3-hop PPR propagation ----
    const float* fin = feat0;
    for (int hop = 0; hop < HOPS; ++hop) {
        hop_kernel<<<EE, 256, 0, stream>>>(src, dst, scores, fin, featB);
        combine_kernel<<<cdiv(nND, 256), 256, 0, stream>>>(featA, featB, feat0, nND);
        fin = featA;
    }

    // ---- output projection + residual ----
    f32_to_bf16_kernel<<<cdiv(nND, 256), 256, 0, stream>>>(featA, featAb, nND);
    wmma_gemm_kernel<2><<<cdiv(groupsNN, 8), 256, 0, stream>>>(
        featAb, WtO, nullptr, ent, rst, nullptr, NN, DD, DD);

    // ---- FFN with residual ----
    ln_bf16_kernel<<<NN, 256, 0, stream>>>(rst, g_ff, b_ff, xnb, NN, NN);
    int groupsMid = (NN / 16) * ((4 * DD) / 64);
    wmma_gemm_kernel<3><<<cdiv(groupsMid, 8), 256, 0, stream>>>(
        xnb, Wt1, b1, nullptr, nullptr, midb, NN, DD, 4 * DD);
    wmma_gemm_kernel<4><<<cdiv(groupsNN, 8), 256, 0, stream>>>(
        midb, Wt2, b2, rst, outp, nullptr, NN, 4 * DD, DD);

    (void)in_sizes; (void)n_in; (void)out_size; (void)ws_size;
}
